// SelfAttention_7980049236170
// MI455X (gfx1250) — compile-verified, matrix operand order checked by reading, not run on hardware
//
#include <hip/hip_runtime.h>
#include <math.h>

typedef __bf16 bf16;
typedef __attribute__((ext_vector_type(16))) __bf16 v16bf;
typedef __attribute__((ext_vector_type(8)))  float  v8f;
typedef __attribute__((ext_vector_type(4))) unsigned int tdm_u4;
typedef __attribute__((ext_vector_type(8))) int          tdm_i8;
typedef __attribute__((ext_vector_type(4))) int          tdm_i4;

#define DMODEL 1024
#define SEQ    2048
#define NBATCH 8

// ---------------------------------------------------------------------------
// WMMA helper: D = A(16x32 bf16) * B(32x16 bf16) + C(16x16 f32)
// ---------------------------------------------------------------------------
__device__ __forceinline__ v8f wmma_bf16(v16bf a, v16bf b, v8f c) {
  return __builtin_amdgcn_wmma_f32_16x16x32_bf16(
      false, a, false, b, (short)0, c, false, false);
}

// 16-bit operand fragment load (ISA 7.12.2 layout). `row` includes (lane&15).
__device__ __forceinline__ v16bf load_frag(const bf16* base, int stride,
                                           int row, int kbase) {
  const int half = (threadIdx.x >> 4) & 1;
  const bf16* p = base + (size_t)row * stride + kbase + half * 8;
  v16bf f;
#pragma unroll
  for (int i = 0; i < 8; ++i) { f[i] = p[i]; f[8 + i] = p[16 + i]; }
  return f;
}

// ---------------------------------------------------------------------------
// Tensor Data Mover: 2D bf16 tile (tile_d0 elems/row x tile_d1 rows) from
// global (row stride `stride0` elems) into LDS at byte offset lds_addr.
// D# layout per CDNA5 ISA 8.3/8.4: group0 = {count, lds_addr, gaddr, type=2},
// group1 = {data_size=1(2B), tensor dims, tile dims, dim0 stride}.
// clang-23 (amdgpu-toolchain) 6-arg builtin form.
// ---------------------------------------------------------------------------
__device__ __forceinline__ void tdm_load_2d_bf16(unsigned lds_addr,
                                                 const bf16* gptr,
                                                 unsigned tile_d0, unsigned tile_d1,
                                                 unsigned tensor_d0, unsigned tensor_d1,
                                                 unsigned stride0) {
  const unsigned long long ga = (unsigned long long)gptr;
  tdm_u4 g0;
  g0.x = 1u;                                        // count=1, user mode
  g0.y = lds_addr;                                  // LDS byte address
  g0.z = (unsigned)(ga & 0xFFFFFFFFu);              // global_addr[31:0]
  g0.w = (unsigned)((ga >> 32) & 0x01FFFFFFu) | (2u << 30);  // addr hi | type=2
  tdm_i8 g1;
  g1[0] = (int)(1u << 16);                          // workgroup_mask=0, data_size=1 (2B)
  g1[1] = (int)((tensor_d0 & 0xFFFFu) << 16);       // [63:48] tensor_dim0 lo16
  g1[2] = (int)(((tensor_d0 >> 16) & 0xFFFFu) |     // [79:64] tensor_dim0 hi16
                ((tensor_d1 & 0xFFFFu) << 16));     // [95:80] tensor_dim1 lo16
  g1[3] = (int)(((tensor_d1 >> 16) & 0xFFFFu) |     // [111:96] tensor_dim1 hi16
                ((tile_d0 & 0xFFFFu) << 16));       // [127:112] tile_dim0
  g1[4] = (int)(tile_d1 & 0xFFFFu);                 // tile_dim1 (tile_dim2=0)
  g1[5] = (int)stride0;                             // tensor_dim0_stride lo32
  g1[6] = 0;                                        // stride hi / dim1 stride lo
  g1[7] = 0;
  const tdm_i4 z4 = {0, 0, 0, 0};
  const tdm_i8 z8 = {0, 0, 0, 0, 0, 0, 0, 0};
  __builtin_amdgcn_tensor_load_to_lds(g0, g1, z4, z4, z8, 0);
}

__device__ __forceinline__ unsigned lds_byte_addr(const void* p) {
  return (unsigned)(unsigned long long)p;           // low 32 bits = LDS offset
}

// ---------------------------------------------------------------------------
// Kernel 1: Q/K/V = x @ W + b (bf16 out).  Software-pipelined: global loads
// for step kk+32 are issued into registers while WMMAs for step kk execute.
// Q,K stored [B*S, D] row-major; V stored transposed [B, D, S].
// ---------------------------------------------------------------------------
__global__ __launch_bounds__(256) void qkv_kernel(
    const float* __restrict__ x,
    const float* __restrict__ Wq, const float* __restrict__ bq,
    const float* __restrict__ Wk, const float* __restrict__ bk,
    const float* __restrict__ Wv, const float* __restrict__ bv,
    bf16* __restrict__ Qh, bf16* __restrict__ Kh, bf16* __restrict__ Vth) {
  __shared__ bf16 Axs[64 * 32];
  __shared__ bf16 Wqs[64 * 32];
  __shared__ bf16 Wks[64 * 32];
  __shared__ bf16 Wvs[64 * 32];

  const int t    = threadIdx.x;
  const int wave = t >> 5;
  const int lane = t & 31;
  const int half = lane >> 4;
  const int ln   = lane & 15;

  const int m0 = blockIdx.y * 64;
  const int n0 = blockIdx.x * 64;
  const int mi = wave & 3;
  const int ch = wave >> 2;

  v8f accQ[2], accK[2], accV[2];
  const v8f vzero = {0.f, 0.f, 0.f, 0.f, 0.f, 0.f, 0.f, 0.f};
#pragma unroll
  for (int nt = 0; nt < 2; ++nt) { accQ[nt] = vzero; accK[nt] = vzero; accV[nt] = vzero; }

  // staging assignments
  const int xr_row = t >> 2;            // 0..63
  const int xr_c   = (t & 3) * 8;       // 0,8,16,24
  const int w_n    = t & 63;            // 0..63
  const int w_kb   = (t >> 6) * 8;      // 0,8,16,24

  float xr[8], wqr[8], wkr[8], wvr[8];
  // preload kk = 0
  {
    const float* src = x + (size_t)(m0 + xr_row) * DMODEL + xr_c;
#pragma unroll
    for (int i = 0; i < 8; ++i) xr[i] = src[i];
#pragma unroll
    for (int i = 0; i < 8; ++i) {
      const size_t gi = (size_t)(w_kb + i) * DMODEL + n0 + w_n;
      wqr[i] = Wq[gi]; wkr[i] = Wk[gi]; wvr[i] = Wv[gi];
    }
  }

  for (int kk = 0; kk < DMODEL; kk += 32) {
    __syncthreads();   // previous step's fragment reads complete
    // commit staged registers -> LDS (f32 -> bf16)
#pragma unroll
    for (int i = 0; i < 8; ++i) Axs[xr_row * 32 + xr_c + i] = (bf16)xr[i];
#pragma unroll
    for (int i = 0; i < 8; ++i) {
      Wqs[w_n * 32 + w_kb + i] = (bf16)wqr[i];
      Wks[w_n * 32 + w_kb + i] = (bf16)wkr[i];
      Wvs[w_n * 32 + w_kb + i] = (bf16)wvr[i];
    }
    __syncthreads();

    // issue next step's global loads (overlap with WMMAs below)
    if (kk + 32 < DMODEL) {
      const float* src = x + (size_t)(m0 + xr_row) * DMODEL + kk + 32 + xr_c;
#pragma unroll
      for (int i = 0; i < 8; ++i) xr[i] = src[i];
#pragma unroll
      for (int i = 0; i < 8; ++i) {
        const size_t gi = (size_t)(kk + 32 + w_kb + i) * DMODEL + n0 + w_n;
        wqr[i] = Wq[gi]; wkr[i] = Wk[gi]; wvr[i] = Wv[gi];
      }
      if (kk + 64 < DMODEL)
        __builtin_prefetch(x + (size_t)(m0 + xr_row) * DMODEL + kk + 64 + xr_c, 0, 0);
    }

    // hoist all fragments, then the WMMA chain
    const v16bf a = load_frag(Axs, 32, mi * 16 + ln, 0);
    const int nr0 = ch * 32 + ln;
    const int nr1 = ch * 32 + 16 + ln;
    const v16bf fq0 = load_frag(Wqs, 32, nr0, 0);
    const v16bf fk0 = load_frag(Wks, 32, nr0, 0);
    const v16bf fv0 = load_frag(Wvs, 32, nr0, 0);
    const v16bf fq1 = load_frag(Wqs, 32, nr1, 0);
    const v16bf fk1 = load_frag(Wks, 32, nr1, 0);
    const v16bf fv1 = load_frag(Wvs, 32, nr1, 0);
    accQ[0] = wmma_bf16(a, fq0, accQ[0]);
    accK[0] = wmma_bf16(a, fk0, accK[0]);
    accV[0] = wmma_bf16(a, fv0, accV[0]);
    accQ[1] = wmma_bf16(a, fq1, accQ[1]);
    accK[1] = wmma_bf16(a, fk1, accK[1]);
    accV[1] = wmma_bf16(a, fv1, accV[1]);
  }

  // epilogue: + bias, store (V transposed)
  const int bI = m0 / SEQ;
#pragma unroll
  for (int nt = 0; nt < 2; ++nt) {
    const int n = n0 + ch * 32 + nt * 16 + ln;
    const float vbq = bq[n], vbk = bk[n], vbv = bv[n];
#pragma unroll
    for (int r = 0; r < 8; ++r) {
      const int m = m0 + mi * 16 + r + 8 * half;
      Qh[(size_t)m * DMODEL + n] = (bf16)(accQ[nt][r] + vbq);
      Kh[(size_t)m * DMODEL + n] = (bf16)(accK[nt][r] + vbk);
      const int sloc = m - bI * SEQ;
      Vth[((size_t)bI * DMODEL + n) * SEQ + sloc] = (bf16)(accV[nt][r] + vbv);
    }
  }
}

// ---------------------------------------------------------------------------
// Kernel 2: flash attention + residual.  K/Q slices staged into ping-pong
// LDS buffers by the Tensor Data Mover (wave0 -> K, wave1 -> Q) while all
// waves compute WMMAs on the other buffer.
// ---------------------------------------------------------------------------
__global__ __launch_bounds__(256) void attn_kernel(
    const bf16* __restrict__ Qh, const bf16* __restrict__ Kh,
    const bf16* __restrict__ Vth, const float* __restrict__ x,
    float* __restrict__ out) {
  __shared__ bf16  Qsl[2][32 * 32];
  __shared__ bf16  Ks [2][128 * 32];
  __shared__ float Ss [32 * 128];
  __shared__ bf16  Ps [32 * 128];
  __shared__ float mS[32], lS[32], alS[32];

  const int t    = threadIdx.x;
  const int wave = t >> 5;
  const int lane = t & 31;
  const int half = lane >> 4;
  const int ln   = lane & 15;

  const int bb = blockIdx.y;
  const int m0 = blockIdx.x * 32;
  const size_t qkBase = (size_t)bb * SEQ;

  const v8f vzero = {0.f, 0.f, 0.f, 0.f, 0.f, 0.f, 0.f, 0.f};
  v8f acc[2][8];
#pragma unroll
  for (int mt = 0; mt < 2; ++mt)
#pragma unroll
    for (int nt = 0; nt < 8; ++nt) acc[mt][nt] = vzero;

  for (int kt = 0; kt < SEQ / 128; ++kt) {
    const int k0 = kt * 128;

    // prefetch next key tile into cache while this one is processed
    if (kt + 1 < SEQ / 128)
      __builtin_prefetch(Kh + (qkBase + k0 + 128 + (t >> 1)) * DMODEL, 0, 0);

    // prologue: DMA first slices into buffer 0
    if (wave == 0)
      tdm_load_2d_bf16(lds_byte_addr(&Ks[0][0]),
                       Kh + (qkBase + k0) * DMODEL,
                       32, 128, DMODEL, SEQ, DMODEL);
    if (wave == 1)
      tdm_load_2d_bf16(lds_byte_addr(&Qsl[0][0]),
                       Qh + (qkBase + m0) * DMODEL,
                       32, 32, DMODEL, SEQ, DMODEL);

    // ---- scores: S[32 x 128] = Q @ K^T over D, TDM double-buffered ----
    v8f sc0 = vzero, sc1 = vzero;
    for (int dd = 0; dd < DMODEL; dd += 32) {
      const int buf = (dd >> 5) & 1;
      if (wave < 2) __builtin_amdgcn_s_wait_tensorcnt(0);
      __syncthreads();                       // buf filled; other buf free
      if (dd + 32 < DMODEL) {                // DMA next slices into other buf
        if (wave == 0)
          tdm_load_2d_bf16(lds_byte_addr(&Ks[buf ^ 1][0]),
                           Kh + (qkBase + k0) * DMODEL + dd + 32,
                           32, 128, DMODEL, SEQ, DMODEL);
        if (wave == 1)
          tdm_load_2d_bf16(lds_byte_addr(&Qsl[buf ^ 1][0]),
                           Qh + (qkBase + m0) * DMODEL + dd + 32,
                           32, 32, DMODEL, SEQ, DMODEL);
      }
      const v16bf a0  = load_frag(Qsl[buf], 32, ln, 0);
      const v16bf a1  = load_frag(Qsl[buf], 32, 16 + ln, 0);
      const v16bf bfr = load_frag(Ks[buf], 32, wave * 16 + ln, 0);
      sc0 = wmma_bf16(a0, bfr, sc0);
      sc1 = wmma_bf16(a1, bfr, sc1);
    }

    // spill scores (each wave owns key columns [wave*16, wave*16+16))
#pragma unroll
    for (int r = 0; r < 8; ++r) {
      Ss[(r + 8 * half) * 128 + wave * 16 + ln]      = sc0[r];
      Ss[(16 + r + 8 * half) * 128 + wave * 16 + ln] = sc1[r];
    }
    __syncthreads();

    // ---- online softmax state update (one thread per query row) ----
    if (t < 32) {
      const float mold = (kt == 0) ? -1e30f : mS[t];
      const float lold = (kt == 0) ? 0.0f   : lS[t];
      float mx = mold;
      for (int c = 0; c < 128; ++c) mx = fmaxf(mx, Ss[t * 128 + c]);
      const float al = __expf(mold - mx);
      float sum = 0.0f;
      for (int c = 0; c < 128; ++c) {
        const float p = __expf(Ss[t * 128 + c] - mx);
        Ps[t * 128 + c] = (bf16)p;
        sum += p;
      }
      mS[t] = mx;
      lS[t] = lold * al + sum;
      alS[t] = al;
    }
    __syncthreads();

    // ---- rescale running output ----
#pragma unroll
    for (int mt = 0; mt < 2; ++mt)
#pragma unroll
      for (int r = 0; r < 8; ++r) {
        const float al = alS[mt * 16 + r + 8 * half];
#pragma unroll
        for (int nt = 0; nt < 8; ++nt) acc[mt][nt][r] *= al;
      }

    // ---- PV: O += P[32 x 128] @ V (per-wave 128-feature chunk) ----
#pragma unroll
    for (int kk = 0; kk < 4; ++kk) {
      const v16bf ap0 = load_frag(Ps, 128, ln, kk * 32);
      const v16bf ap1 = load_frag(Ps, 128, 16 + ln, kk * 32);
#pragma unroll
      for (int nt = 0; nt < 8; ++nt) {
        const size_t vrow = (size_t)bb * DMODEL + wave * 128 + nt * 16 + ln;
        const bf16* p = Vth + vrow * SEQ + k0 + kk * 32 + half * 8;
        v16bf bvf;
#pragma unroll
        for (int i = 0; i < 8; ++i) { bvf[i] = p[i]; bvf[8 + i] = p[16 + i]; }
        acc[0][nt] = wmma_bf16(ap0, bvf, acc[0][nt]);
        acc[1][nt] = wmma_bf16(ap1, bvf, acc[1][nt]);
      }
    }
  }

  // ---- epilogue: out = O / l + x ----
#pragma unroll
  for (int mt = 0; mt < 2; ++mt)
#pragma unroll
    for (int r = 0; r < 8; ++r) {
      const int row = mt * 16 + r + 8 * half;
      const float inv = 1.0f / lS[row];
      const size_t gRow = (qkBase + m0 + row) * DMODEL;
#pragma unroll
      for (int nt = 0; nt < 8; ++nt) {
        const int n = wave * 128 + nt * 16 + ln;
        out[gRow + n] = acc[mt][nt][r] * inv + x[gRow + n];
      }
    }
}

// ---------------------------------------------------------------------------
extern "C" void kernel_launch(void* const* d_in, const int* in_sizes, int n_in,
                              void* d_out, int out_size, void* d_ws, size_t ws_size,
                              hipStream_t stream) {
  const float* x  = (const float*)d_in[0];
  const float* Wq = (const float*)d_in[1];
  const float* bq = (const float*)d_in[2];
  const float* Wk = (const float*)d_in[3];
  const float* bk = (const float*)d_in[4];
  const float* Wv = (const float*)d_in[5];
  const float* bv = (const float*)d_in[6];
  float* out = (float*)d_out;

  const size_t tensorElems = (size_t)NBATCH * SEQ * DMODEL;
  bf16* Qh  = (bf16*)d_ws;
  bf16* Kh  = Qh + tensorElems;
  bf16* Vth = Kh + tensorElems;   // [B, D, S]

  dim3 g1(DMODEL / 64, (NBATCH * SEQ) / 64);
  qkv_kernel<<<g1, 256, 0, stream>>>(x, Wq, bq, Wk, bk, Wv, bv, Qh, Kh, Vth);

  dim3 g2(SEQ / 32, NBATCH);
  attn_kernel<<<g2, 256, 0, stream>>>(Qh, Kh, Vth, x, out);
}
